// LSTM_78589311582428
// MI455X (gfx1250) — matrix operand order repaired and checked by reading, not compiled, not run on hardware
//
#include <hip/hip_runtime.h>
#include <hip/hip_bf16.h>

// ---------------------------------------------------------------------------
// LSTM on gfx1250: bf16 WMMA recurrent steps, fused gate elementwise,
// distance-2 register software pipeline + WGP-scope global_prefetch.
//
// Shapes: B=256, T=512, D=256, H=1024, C=10.  K_total = D+H = 1280.
// Gate order (columns of the 4H axis): i, f, g, o.
// ---------------------------------------------------------------------------

typedef __attribute__((ext_vector_type(16))) __bf16 bf16x16;
typedef __attribute__((ext_vector_type(8)))  __bf16 bf16x8;
typedef __attribute__((ext_vector_type(8)))  float  floatx8;

#define BATCH   256
#define SEQLEN  512
#define DIN     256
#define HID     1024
#define NCLS    10
#define KTOT    1280           // DIN + HID
#define KTILES  40             // KTOT / 32
#define NTOT    4096           // 4*HID

// ---------------------------------------------------------------------------
// Pack the 4 gate input+recurrent weight matrices (fp32) into one bf16 matrix
// of shape [KTOT, NTOT], stored in WMMA B-fragment order:
//   packed[((nTile*KTILES + kTile)*32 + lane)*16 + j]
//     = W[kTile*32 + (lane>=16 ? 8 : 0) + (j&7) + (j>=8 ? 16 : 0)][nTile*16 + (lane&15)]
// so each lane of a wave reads its 16 bf16 B-fragment values contiguously.
// ---------------------------------------------------------------------------
__global__ void lstm_pack_w(const float* __restrict__ Wix, const float* __restrict__ Wih,
                            const float* __restrict__ Wfx, const float* __restrict__ Wfh,
                            const float* __restrict__ Wgx, const float* __restrict__ Wgh,
                            const float* __restrict__ Wox, const float* __restrict__ Woh,
                            __bf16* __restrict__ packed) {
    const int idx = blockIdx.x * blockDim.x + threadIdx.x;   // < 256*40*32*16
    const int j    = idx & 15;
    const int lane = (idx >> 4) & 31;
    const int tile = idx >> 9;            // nTile*KTILES + kTile
    const int kt   = tile % KTILES;
    const int nt   = tile / KTILES;       // 0..255
    const int n    = nt * 16 + (lane & 15);          // 0..4095
    const int k    = kt * 32 + ((lane >> 4) << 3) + (j & 7) + ((j >> 3) << 4);
    const int gate = n >> 10;             // 0:i 1:f 2:g 3:o
    const int col  = n & 1023;
    const float* Wx;
    const float* Wh;
    switch (gate) {
        case 0:  Wx = Wix; Wh = Wih; break;
        case 1:  Wx = Wfx; Wh = Wfh; break;
        case 2:  Wx = Wgx; Wh = Wgh; break;
        default: Wx = Wox; Wh = Woh; break;
    }
    const float v = (k < DIN) ? Wx[(size_t)k * HID + col]
                              : Wh[(size_t)(k - DIN) * HID + col];
    packed[idx] = (__bf16)v;
}

// Concatenated bias [4096] fp32, gate order i,f,g,o.
__global__ void lstm_pack_bias(const float* __restrict__ bi, const float* __restrict__ bf_,
                               const float* __restrict__ bg, const float* __restrict__ bo,
                               float* __restrict__ bias) {
    const int n = blockIdx.x * blockDim.x + threadIdx.x;     // < 4096
    const int col = n & 1023;
    float v;
    switch (n >> 10) {
        case 0:  v = bi[col];  break;
        case 1:  v = bf_[col]; break;
        case 2:  v = bg[col];  break;
        default: v = bo[col];  break;
    }
    bias[n] = v;
}

// x[B,T,D] fp32 -> xT[T,B,D] bf16 (step t reads one contiguous 256x256 tile)
__global__ void lstm_pack_x(const float* __restrict__ x, __bf16* __restrict__ xT) {
    const int idx = blockIdx.x * blockDim.x + threadIdx.x;   // < T*B*D
    const int d = idx & (DIN - 1);
    const int b = (idx >> 8) & (BATCH - 1);
    const int t = idx >> 16;
    xT[idx] = (__bf16)x[((size_t)b * SEQLEN + t) * DIN + d];
}

// h0 = 0 (bf16), c0 = 0 (fp32)
__global__ void lstm_init_state(__bf16* __restrict__ h0, float* __restrict__ c0) {
    const int idx = blockIdx.x * blockDim.x + threadIdx.x;   // < B*H
    h0[idx] = (__bf16)0.0f;
    c0[idx] = 0.0f;
}

__device__ __forceinline__ float sigf(float x) {
    return 1.0f / (1.0f + __expf(-x));
}

// WGP-scope prefetch (scope field omitted -> 0 = WGP: pulls into all cache
// levels, i.e. the L2 -> WGP$ hop we actually need for the resident weights).
__device__ __forceinline__ void pf_wgp(const void* p) {
    asm volatile("global_prefetch_b8 %0, off" :: "v"(p));
}

// Per-lane A-fragment address (16-bit A 16x32 layout): lane holds row
// m0+lane%16, chunks at K offsets ksel and ksel+16 within the 32-wide tile.
__device__ __forceinline__ const __bf16* addrA(const __bf16* __restrict__ xt,
                                               const __bf16* __restrict__ h_in,
                                               int rowm, int ksel, int kt) {
    const int k0 = kt * 32;
    if (k0 < DIN) return xt   + (size_t)rowm * DIN + k0 + ksel;
    return              h_in + (size_t)rowm * HID + (k0 - DIN) + ksel;
}

__device__ __forceinline__ bf16x16 loadA(const __bf16* __restrict__ xt,
                                         const __bf16* __restrict__ h_in,
                                         int rowm, int ksel, int kt) {
    const __bf16* ap = addrA(xt, h_in, rowm, ksel, kt);
    union { bf16x16 v; bf16x8 p[2]; } u;
    u.p[0] = *(const bf16x8*)(ap);
    u.p[1] = *(const bf16x8*)(ap + 16);
    return u.v;
}

// ---------------------------------------------------------------------------
// One recurrent step:
//   gates[256,4096] = [x_t | h_in] @ Wpacked + bias    (bf16 WMMA, f32 acc)
//   i,f,g,o = sigmoid(gates);  c = g*i + c*f;  h_out = tanh(c)*o
//
// 128 threads = 4 waves.  All 4 waves share ONE 16-wide hidden column tile
// (4 gate B-fragments per K-step, WGP$-shared); wave w owns batch rows
// [bx*64 + w*16, +16).  Four fragment sets give a distance-2 software
// pipeline: loads for kt+2/kt+3 are in flight while kt/kt+1 compute.
// Grid: (BATCH/64, HID/16) = (4, 64) = 256 blocks.
// ---------------------------------------------------------------------------
__global__ __launch_bounds__(128) void lstm_step(
    const __bf16* __restrict__ xt,      // [256, 256]  row-major bf16 (time t)
    const __bf16* __restrict__ h_in,    // [256, 1024] row-major bf16
    const __bf16* __restrict__ Wp,      // packed B fragments
    const float*  __restrict__ bias,    // [4096]
    float*        __restrict__ c,       // [256, 1024] fp32 (in/out)
    __bf16*       __restrict__ h_out)   // [256, 1024] bf16
{
    const int lane = threadIdx.x & 31;
    const int wid  = threadIdx.x >> 5;
    const int m0 = blockIdx.x * 64 + wid * 16;       // batch row base (per wave)
    const int n0 = blockIdx.y * 16;                  // hidden column base (per block)

    floatx8 acc[4];
#pragma unroll
    for (int g = 0; g < 4; ++g) acc[g] = (floatx8)0.0f;

    const int rowm = m0 + (lane & 15);
    const int ksel = (lane >> 4) << 3;               // 0 or 8

    // Per-gate packed-B lane pointers: nTile(g) = g*64 + n0/16.
    const __bf16* bptr[4];
#pragma unroll
    for (int g = 0; g < 4; ++g) {
        const int ntile = g * (HID / 16) + (n0 >> 4);
        bptr[g] = Wp + ((size_t)ntile * KTILES) * 512 + (size_t)lane * 16;
    }

    // Load one fragment set (A + 4 gate B fragments) for K-tile kt.
    auto LD = [&](int kt, bf16x16& a, bf16x16* b) {
        a = loadA(xt, h_in, rowm, ksel, kt);
#pragma unroll
        for (int g = 0; g < 4; ++g)
            b[g] = *(const bf16x16*)(bptr[g] + (size_t)kt * 512);
    };
    auto MM = [&](const bf16x16& a, const bf16x16* b) {
#pragma unroll
        for (int g = 0; g < 4; ++g)
            acc[g] = __builtin_amdgcn_wmma_f32_16x16x32_bf16(
                false, a, false, b[g], (short)0, acc[g], false, false);
    };

    // --- distance-2 software pipeline over KTILES (=40) ---
    bf16x16 aS0, aS1, aS2, aS3;
    bf16x16 bS0[4], bS1[4], bS2[4], bS3[4];

    LD(0, aS0, bS0);
    LD(1, aS1, bS1);

    for (int kt = 0; kt < KTILES; kt += 4) {
        const int k2 = (kt + 2 < KTILES) ? kt + 2 : KTILES - 1;
        const int k3 = (kt + 3 < KTILES) ? kt + 3 : KTILES - 1;
        LD(k2, aS2, bS2);
        LD(k3, aS3, bS3);

        // Stream-prefetch ~8 K-tiles ahead into WGP$ (block B working set is
        // 160KB/step, exceeding the 64KB WGP$, so it streams from L2).
        {
            const int kp = (kt + 8 < KTILES) ? kt + 8 : KTILES - 1;
#pragma unroll
            for (int g = 0; g < 4; ++g)
                pf_wgp((const void*)(bptr[g] + (size_t)kp * 512));
            pf_wgp((const void*)addrA(xt, h_in, rowm, ksel, kp));
        }

        MM(aS0, bS0);
        MM(aS1, bS1);

        const int k4 = (kt + 4 < KTILES) ? kt + 4 : KTILES - 1;
        const int k5 = (kt + 5 < KTILES) ? kt + 5 : KTILES - 1;
        LD(k4, aS0, bS0);
        LD(k5, aS1, bS1);

        MM(aS2, bS2);
        MM(aS3, bS3);
    }

    // Elementwise: C/D layout -> vgpr r, lane l : M = m0 + r + (l>=16?8:0),
    //                                            N = n0 + (l&15)
    const int mbase = m0 + ((lane >> 4) << 3);
    const int ncol  = n0 + (lane & 15);
    const float bi_ = bias[ncol];
    const float bf_ = bias[HID + ncol];
    const float bg_ = bias[2 * HID + ncol];
    const float bo_ = bias[3 * HID + ncol];
#pragma unroll
    for (int r = 0; r < 8; ++r) {
        const int m = mbase + r;
        const float i_ = sigf(acc[0][r] + bi_);
        const float f_ = sigf(acc[1][r] + bf_);
        const float g_ = sigf(acc[2][r] + bg_);
        const float o_ = sigf(acc[3][r] + bo_);
        const size_t off = (size_t)m * HID + ncol;
        const float cn = g_ * i_ + c[off] * f_;
        c[off] = cn;
        h_out[off] = (__bf16)(tanhf(cn) * o_);
    }
}

// out[256,10] = h_T(bf16) @ Wph(fp32 [1024,10]) + bp
__global__ void lstm_final_proj(const __bf16* __restrict__ h,
                                const float* __restrict__ Wph,
                                const float* __restrict__ bp,
                                float* __restrict__ out) {
    const int idx = blockIdx.x * blockDim.x + threadIdx.x;
    if (idx >= BATCH * NCLS) return;
    const int b  = idx / NCLS;
    const int cl = idx % NCLS;
    float s = bp[cl];
    const __bf16* hr = h + (size_t)b * HID;
#pragma unroll 4
    for (int k = 0; k < HID; ++k)
        s += (float)hr[k] * Wph[(size_t)k * NCLS + cl];
    out[idx] = s;
}

// ---------------------------------------------------------------------------
extern "C" void kernel_launch(void* const* d_in, const int* in_sizes, int n_in,
                              void* d_out, int out_size, void* d_ws, size_t ws_size,
                              hipStream_t stream) {
    const float* x   = (const float*)d_in[0];
    const float* Wgx = (const float*)d_in[1];
    const float* Wgh = (const float*)d_in[2];
    const float* bg  = (const float*)d_in[3];
    const float* Wix = (const float*)d_in[4];
    const float* Wih = (const float*)d_in[5];
    const float* bi  = (const float*)d_in[6];
    const float* Wfx = (const float*)d_in[7];
    const float* Wfh = (const float*)d_in[8];
    const float* bf_ = (const float*)d_in[9];
    const float* Wox = (const float*)d_in[10];
    const float* Woh = (const float*)d_in[11];
    const float* bo  = (const float*)d_in[12];
    const float* Wph = (const float*)d_in[13];
    const float* bp  = (const float*)d_in[14];
    float* out = (float*)d_out;

    // Workspace layout (all offsets 256B-aligned by construction).
    char* ws = (char*)d_ws;
    __bf16* packedW = (__bf16*)(ws);                              // 10,485,760 B
    float*  bias    = (float*)(ws + 10485760);                    //     16,384 B
    __bf16* xT      = (__bf16*)(ws + 10502144);                   // 67,108,864 B
    __bf16* h0      = (__bf16*)(ws + 77611008);                   //    524,288 B
    __bf16* h1      = (__bf16*)(ws + 78135296);                   //    524,288 B
    float*  cbuf    = (float*)(ws + 78659584);                    //  1,048,576 B
    // total: 79,708,160 bytes

    // --- pack (runs every call; deterministic) ---
    {
        const int n = 256 * KTILES * 32 * 16;                     // 5,242,880
        lstm_pack_w<<<n / 256, 256, 0, stream>>>(Wix, Wih, Wfx, Wfh,
                                                 Wgx, Wgh, Wox, Woh, packedW);
    }
    lstm_pack_bias<<<NTOT / 256, 256, 0, stream>>>(bi, bf_, bg, bo, bias);
    {
        const int n = SEQLEN * BATCH * DIN;                       // 33,554,432
        lstm_pack_x<<<n / 256, 256, 0, stream>>>(x, xT);
    }
    lstm_init_state<<<(BATCH * HID) / 256, 256, 0, stream>>>(h0, cbuf);

    // --- recurrent scan: 512 steps, ping-pong h buffers ---
    const dim3 sgrid(BATCH / 64, HID / 16);                       // (4,64) = 256 blocks
    for (int t = 0; t < SEQLEN; ++t) {
        const __bf16* hin = (t & 1) ? h1 : h0;
        __bf16*       hout = (t & 1) ? h0 : h1;
        lstm_step<<<sgrid, 128, 0, stream>>>(xT + (size_t)t * BATCH * DIN,
                                             hin, packedW, bias, cbuf, hout);
    }
    // After 512 steps (even), final h lives in h0.

    lstm_final_proj<<<(BATCH * NCLS + 255) / 256, 256, 0, stream>>>(h0, Wph, bp, out);
    (void)in_sizes; (void)n_in; (void)out_size; (void)ws_size;
}
